// Decoder_5574867550758
// MI455X (gfx1250) — compile-verified, matrix operand order checked by reading, not run on hardware
//
#include <hip/hip_runtime.h>
#include <hip/hip_bf16.h>

// ---------------------------------------------------------------------------
// CDNA5 (gfx1250) decoder, MI455X. ~400 GFLOP of dense GEMM vs ~25us of HBM
// traffic at 23.3 TB/s => compute-bound => everything goes through
// v_wmma_f32_16x16x32_bf16 (wave32). All GEMM operands are pre-packed to
// bf16 [N][K] row-major so every LDS tile stage is a pure async DMA
// (global_load_async_to_lds_b128, ASYNCcnt) with ZERO VALU in the staging
// path -- conversion/transpose work is hoisted into cheap bandwidth-bound
// pre-pass kernels.
// ---------------------------------------------------------------------------

typedef __attribute__((ext_vector_type(16))) __bf16 v16bf;
typedef __attribute__((ext_vector_type(8)))  __bf16 v8bf;
typedef __attribute__((ext_vector_type(8)))  float  v8f;

static constexpr int T_DIM = 4096;   // training_step == T (full sequence)
static constexpr int S_DIM = 4096;
static constexpr int D_DIM = 512;
static constexpr int F_DIM = 2048;
static constexpr int L_DIM = 4;

static constexpr int BM = 128, BN = 128, BK = 32;
static constexpr int LDSK = 40;      // padded K pitch (16B-aligned rows, spreads banks)

static constexpr float LN_EPS  = 1e-6f;
static constexpr float NEG_BIG = 1e9f;

// ---------------- async VMEM -> LDS staging (gfx1250) -----------------------
// Copies 32 contiguous bytes per thread: global row-major [rows][ld] bf16 tile
// into LDS [128][LDSK]. 256 threads cover a 128x32 bf16 tile (8 KB).
// INST_OFFSET is added to BOTH the LDS and global addresses (ISA 10.x).

__device__ __forceinline__ void stage_async_tile(const __bf16* src, size_t ld,
                                                 size_t row0, size_t col0,
                                                 unsigned lds_base, int tid) {
  const int r  = tid >> 1;
  const int c0 = (tid & 1) * 16;
  const __bf16* g = src + (row0 + (size_t)r) * ld + col0 + (size_t)c0;
  const unsigned l = lds_base + (unsigned)(r * LDSK + c0) * 2u;
  asm volatile("global_load_async_to_lds_b128 %0, %1, off"
               :: "v"(l), "v"(g) : "memory");
  asm volatile("global_load_async_to_lds_b128 %0, %1, off offset:16"
               :: "v"(l), "v"(g) : "memory");
}

__device__ __forceinline__ void wait_async_zero() {
  asm volatile("s_wait_asynccnt 0x0" ::: "memory");
}

// ---------------- WMMA GEMM --------------------------------------------------
// C[M,N] = A[M,K] @ B^T, A row-major [M][K] bf16, B row-major [N][K] bf16.
// EPI: 0 = f32 store, 1 = bf16 store, 2 = relu->bf16 store,
//      3 = R + C -> f32 store (residual), 4 = bf16 TRANSPOSED store (C^T).
template <int EPI>
__global__ __launch_bounds__(256) void gemm_wmma(const __bf16* __restrict__ A,
                                                 const __bf16* __restrict__ B,
                                                 const float* R, void* Cout,
                                                 int M, int N, int K) {
  __shared__ __bf16 As[2][BM][LDSK];
  __shared__ __bf16 Bs[2][BN][LDSK];

  const int tid  = threadIdx.x;
  const size_t m0 = (size_t)blockIdx.y * BM;
  const size_t n0 = (size_t)blockIdx.x * BN;
  const int ksteps = K / BK;

  const int wave  = tid >> 5;
  const int lane  = tid & 31;
  const int wm    = wave >> 1;       // 0..3 : M direction
  const int wn    = wave & 1;        // 0..1 : N direction
  const int lrow  = lane & 15;
  const int khalf = lane >> 4;       // lanes 16-31 hold K 8-15 / 24-31

  const unsigned asb[2] = { (unsigned)(uintptr_t)&As[0][0][0],
                            (unsigned)(uintptr_t)&As[1][0][0] };
  const unsigned bsb[2] = { (unsigned)(uintptr_t)&Bs[0][0][0],
                            (unsigned)(uintptr_t)&Bs[1][0][0] };

  v8f acc[2][4];
#pragma unroll
  for (int i = 0; i < 2; ++i)
#pragma unroll
    for (int j = 0; j < 4; ++j)
#pragma unroll
      for (int e = 0; e < 8; ++e) acc[i][j][e] = 0.0f;

  auto stage = [&](int s, int buf) {
    stage_async_tile(A, (size_t)K, m0, (size_t)s * BK, asb[buf], tid);
    stage_async_tile(B, (size_t)K, n0, (size_t)s * BK, bsb[buf], tid);
  };

  stage(0, 0);

  for (int s = 0; s < ksteps; ++s) {
    wait_async_zero();       // this wave's async LDS writes are complete
    __syncthreads();         // every wave's writes are complete
    const int cur = s & 1;
    if (s + 1 < ksteps) stage(s + 1, cur ^ 1);
    if (s + 4 < ksteps) {    // pull K-step s+4 toward L2 (global_prefetch_b8)
      __builtin_prefetch(A + (m0 + (size_t)(tid >> 1)) * (size_t)K +
                             (size_t)(s + 4) * BK, 0, 1);
      __builtin_prefetch(B + (n0 + (size_t)(tid >> 1)) * (size_t)K +
                             (size_t)(s + 4) * BK, 0, 1);
    }

    // Fragment loads per documented 16-bit A layout: lane<16 -> K 0-7 & 16-23,
    // lane>=16 -> K 8-15 & 24-31; two ds_load_b128 per fragment.
    v16bf afr[2], bfr[4];
#pragma unroll
    for (int tm = 0; tm < 2; ++tm) {
      const int row = wm * 32 + tm * 16 + lrow;
      v8bf lo = *reinterpret_cast<const v8bf*>(&As[cur][row][khalf * 8]);
      v8bf hi = *reinterpret_cast<const v8bf*>(&As[cur][row][16 + khalf * 8]);
#pragma unroll
      for (int i = 0; i < 8; ++i) { afr[tm][i] = lo[i]; afr[tm][8+i] = hi[i]; }
    }
#pragma unroll
    for (int tn = 0; tn < 4; ++tn) {
      const int col = wn * 64 + tn * 16 + lrow;
      v8bf lo = *reinterpret_cast<const v8bf*>(&Bs[cur][col][khalf * 8]);
      v8bf hi = *reinterpret_cast<const v8bf*>(&Bs[cur][col][16 + khalf * 8]);
#pragma unroll
      for (int i = 0; i < 8; ++i) { bfr[tn][i] = lo[i]; bfr[tn][8+i] = hi[i]; }
    }

#pragma unroll
    for (int tm = 0; tm < 2; ++tm)
#pragma unroll
      for (int tn = 0; tn < 4; ++tn)
        acc[tm][tn] = __builtin_amdgcn_wmma_f32_16x16x32_bf16(
            false, afr[tm], false, bfr[tn], (short)0, acc[tm][tn], false, false);
  }

  // Epilogue: C layout -- VGPR j holds M=j (lanes 0-15) / M=j+8 (lanes 16-31).
#pragma unroll
  for (int tm = 0; tm < 2; ++tm)
#pragma unroll
    for (int tn = 0; tn < 4; ++tn)
#pragma unroll
      for (int j = 0; j < 8; ++j) {
        const size_t row = m0 + (size_t)(wm * 32 + tm * 16 + j + (khalf ? 8 : 0));
        const size_t col = n0 + (size_t)(wn * 64 + tn * 16 + lrow);
        float v = acc[tm][tn][j];
        if constexpr (EPI == 0) {
          reinterpret_cast<float*>(Cout)[row * (size_t)N + col] = v;
        } else if constexpr (EPI == 1) {
          reinterpret_cast<__bf16*>(Cout)[row * (size_t)N + col] = (__bf16)v;
        } else if constexpr (EPI == 2) {
          reinterpret_cast<__bf16*>(Cout)[row * (size_t)N + col] =
              (__bf16)(v > 0.0f ? v : 0.0f);
        } else if constexpr (EPI == 3) {
          const size_t idx = row * (size_t)N + col;
          reinterpret_cast<float*>(Cout)[idx] = R[idx] + v;
        } else {  // EPI == 4: store C^T as bf16  (Cout has shape [N][M])
          reinterpret_cast<__bf16*>(Cout)[col * (size_t)M + row] = (__bf16)v;
        }
      }
}

// ---------------- operand pre-pack kernels ----------------------------------

// Wt[n][k] = (bf16) W[k][n]   -- 32x32 LDS-tiled transpose-convert.
__global__ __launch_bounds__(256) void transpose_bf16_kernel(
    const float* __restrict__ W, __bf16* __restrict__ Wt, int K, int N) {
  __shared__ float tile[32][33];
  const int n0 = blockIdx.x * 32;
  const int k0 = blockIdx.y * 32;
  const int tx = threadIdx.x & 31;
  const int ty = threadIdx.x >> 5;           // 0..7
#pragma unroll
  for (int i = 0; i < 4; ++i) {
    const int k = ty + i * 8;
    tile[k][tx] = W[(size_t)(k0 + k) * N + n0 + tx];
  }
  __syncthreads();
#pragma unroll
  for (int i = 0; i < 4; ++i) {
    const int n = ty + i * 8;
    Wt[(size_t)(n0 + n) * K + k0 + tx] = (__bf16)tile[tx][n];
  }
}

// Y = (bf16) X, flat.
__global__ __launch_bounds__(256) void cvt_bf16_kernel(const float* __restrict__ X,
                                                       __bf16* __restrict__ Y) {
  const size_t i = ((size_t)blockIdx.x * 256 + threadIdx.x) * 4;
  const float4 f = *reinterpret_cast<const float4*>(X + i);
  Y[i + 0] = (__bf16)f.x;
  Y[i + 1] = (__bf16)f.y;
  Y[i + 2] = (__bf16)f.z;
  Y[i + 3] = (__bf16)f.w;
}

// ---------------- LayerNorm (ddof=1), bf16 output ---------------------------

__global__ __launch_bounds__(256) void ln_bf16_kernel(const float* __restrict__ X,
                                                      __bf16* __restrict__ H) {
  __shared__ float red[256];
  const int row = blockIdx.x, tid = threadIdx.x;
  const float* xr = X + (size_t)row * D_DIM;
  float x0 = xr[tid], x1 = xr[tid + 256];
  red[tid] = x0 + x1;
  __syncthreads();
  for (int off = 128; off > 0; off >>= 1) {
    if (tid < off) red[tid] += red[tid + off];
    __syncthreads();
  }
  const float mean = red[0] * (1.0f / (float)D_DIM);
  __syncthreads();
  const float d0 = x0 - mean, d1 = x1 - mean;
  red[tid] = d0 * d0 + d1 * d1;
  __syncthreads();
  for (int off = 128; off > 0; off >>= 1) {
    if (tid < off) red[tid] += red[tid + off];
    __syncthreads();
  }
  const float s = sqrtf(red[0] * (1.0f / (float)(D_DIM - 1))) + LN_EPS;
  __bf16* hr = H + (size_t)row * D_DIM;
  hr[tid]       = (__bf16)(d0 / s);
  hr[tid + 256] = (__bf16)(d1 / s);
}

// ---------------- masked softmax: f32 scores -> bf16 probabilities ----------
// ids != nullptr => self-attn: mask[row][ids[col]] ; else cross: mask[row][col]

__global__ __launch_bounds__(256) void softmax_mask_kernel(
    const float* __restrict__ Sc, const float* __restrict__ mask,
    const int* ids, __bf16* __restrict__ P, float scale) {
  __shared__ float red[256];
  const int row = blockIdx.x, tid = threadIdx.x;
  const float* srow = Sc + (size_t)row * S_DIM;
  const float* mrow = mask + (size_t)row * S_DIM;
  float v[16];
  float mx = -3.4e38f;
#pragma unroll
  for (int i = 0; i < 16; ++i) {
    const int c = tid + 256 * i;
    const float m = ids ? mrow[ids[c]] : mrow[c];
    const float val = srow[c] * scale + (m - 1.0f) * NEG_BIG;
    v[i] = val;
    mx = fmaxf(mx, val);
  }
  red[tid] = mx;
  __syncthreads();
  for (int off = 128; off > 0; off >>= 1) {
    if (tid < off) red[tid] = fmaxf(red[tid], red[tid + off]);
    __syncthreads();
  }
  mx = red[0];
  __syncthreads();
  float sum = 0.0f;
#pragma unroll
  for (int i = 0; i < 16; ++i) { v[i] = __expf(v[i] - mx); sum += v[i]; }
  red[tid] = sum;
  __syncthreads();
  for (int off = 128; off > 0; off >>= 1) {
    if (tid < off) red[tid] += red[tid + off];
    __syncthreads();
  }
  const float inv = 1.0f / red[0];
  __bf16* pr = P + (size_t)row * S_DIM;
#pragma unroll
  for (int i = 0; i < 16; ++i) pr[tid + 256 * i] = (__bf16)(v[i] * inv);
}

// ---------------- final: out = alpha * norm(x) + gamma ----------------------

__global__ __launch_bounds__(256) void final_kernel(const float* __restrict__ X,
                                                    const float* alpha,
                                                    const float* gamma,
                                                    float* __restrict__ out) {
  __shared__ float red[256];
  const int row = blockIdx.x, tid = threadIdx.x;
  const float* xr = X + (size_t)row * D_DIM;
  float x0 = xr[tid], x1 = xr[tid + 256];
  red[tid] = x0 + x1;
  __syncthreads();
  for (int off = 128; off > 0; off >>= 1) {
    if (tid < off) red[tid] += red[tid + off];
    __syncthreads();
  }
  const float mean = red[0] * (1.0f / (float)D_DIM);
  __syncthreads();
  const float d0 = x0 - mean, d1 = x1 - mean;
  red[tid] = d0 * d0 + d1 * d1;
  __syncthreads();
  for (int off = 128; off > 0; off >>= 1) {
    if (tid < off) red[tid] += red[tid + off];
    __syncthreads();
  }
  const float s = sqrtf(red[0] * (1.0f / (float)(D_DIM - 1))) + LN_EPS;
  const float a = alpha[0], g = gamma[0];
  float* orow = out + (size_t)row * D_DIM;
  orow[tid]       = a * (d0 / s) + g;
  orow[tid + 256] = a * (d1 / s) + g;
}

// ---------------- host orchestration ----------------------------------------

extern "C" void kernel_launch(void* const* d_in, const int* in_sizes, int n_in,
                              void* d_out, int out_size, void* d_ws, size_t ws_size,
                              hipStream_t stream) {
  (void)in_sizes; (void)n_in; (void)out_size; (void)ws_size;

  const float* x_in  = (const float*)d_in[0];   // [T,D]
  const float* enc   = (const float*)d_in[1];   // [S,D]
  const float* cmask = (const float*)d_in[2];   // [T,S]
  const float* Wq = (const float*)d_in[3];
  const float* Wk = (const float*)d_in[4];
  const float* Wv = (const float*)d_in[5];
  const float* Wo = (const float*)d_in[6];
  const float* Cq = (const float*)d_in[7];
  const float* Ck = (const float*)d_in[8];
  const float* Cv = (const float*)d_in[9];
  const float* Co = (const float*)d_in[10];
  const float* W1 = (const float*)d_in[11];
  const float* W2 = (const float*)d_in[12];
  const float* alpha = (const float*)d_in[13];
  const float* gamma = (const float*)d_in[14];
  const int*   ids   = (const int*)d_in[15];

  // Workspace carve-out (~150 MB).
  char* w = (char*)d_ws;
  auto carve = [&](size_t bytes) {
    void* p = (void*)w;
    w += (bytes + 255) & ~(size_t)255;
    return p;
  };
  float*  x     = (float*)carve((size_t)T_DIM * D_DIM * sizeof(float));
  __bf16* hbf   = (__bf16*)carve((size_t)T_DIM * D_DIM * sizeof(__bf16));
  __bf16* encb  = (__bf16*)carve((size_t)S_DIM * D_DIM * sizeof(__bf16));
  __bf16* Qb    = (__bf16*)carve((size_t)T_DIM * D_DIM * sizeof(__bf16));
  __bf16* Kb    = (__bf16*)carve((size_t)S_DIM * D_DIM * sizeof(__bf16));
  __bf16* Vt    = (__bf16*)carve((size_t)D_DIM * S_DIM * sizeof(__bf16)); // [D][S]
  __bf16* AVb   = (__bf16*)carve((size_t)T_DIM * D_DIM * sizeof(__bf16));
  __bf16* Hf    = (__bf16*)carve((size_t)T_DIM * F_DIM * sizeof(__bf16));
  __bf16* Wt    = (__bf16*)carve((size_t)F_DIM * D_DIM * sizeof(__bf16)); // scratch W^T
  float*  Sf    = (float*)carve((size_t)T_DIM * S_DIM * sizeof(float));
  __bf16* Pb    = (__bf16*)carve((size_t)T_DIM * S_DIM * sizeof(__bf16));

  const float scale = 1.0f / sqrtf((float)D_DIM);
  const size_t DD = (size_t)D_DIM * D_DIM;
  const size_t DF = (size_t)D_DIM * F_DIM;

  hipMemcpyAsync(x, x_in, (size_t)T_DIM * D_DIM * sizeof(float),
                 hipMemcpyDeviceToDevice, stream);
  cvt_bf16_kernel<<<(S_DIM * D_DIM) / 1024, 256, 0, stream>>>(enc, encb);

  auto gdim  = [](int M, int N) { return dim3((unsigned)(N / BN), (unsigned)(M / BM)); };
  auto tdim  = [](int K, int N) { return dim3((unsigned)(N / 32), (unsigned)(K / 32)); };
  auto convT = [&](const float* Wp, int K, int N) {
    transpose_bf16_kernel<<<tdim(K, N), 256, 0, stream>>>(Wp, Wt, K, N);
  };

  for (int l = 0; l < L_DIM; ++l) {
    // ---- self attention ----
    ln_bf16_kernel<<<T_DIM, 256, 0, stream>>>(x, hbf);
    convT(Wq + (size_t)l * DD, D_DIM, D_DIM);
    gemm_wmma<1><<<gdim(T_DIM, D_DIM), 256, 0, stream>>>(hbf, Wt, nullptr, Qb,
                                                         T_DIM, D_DIM, D_DIM);
    convT(Wk + (size_t)l * DD, D_DIM, D_DIM);
    gemm_wmma<1><<<gdim(T_DIM, D_DIM), 256, 0, stream>>>(hbf, Wt, nullptr, Kb,
                                                         T_DIM, D_DIM, D_DIM);
    convT(Wv + (size_t)l * DD, D_DIM, D_DIM);
    gemm_wmma<4><<<gdim(T_DIM, D_DIM), 256, 0, stream>>>(hbf, Wt, nullptr, Vt,
                                                         T_DIM, D_DIM, D_DIM);
    gemm_wmma<0><<<gdim(T_DIM, S_DIM), 256, 0, stream>>>(Qb, Kb, nullptr, Sf,
                                                         T_DIM, S_DIM, D_DIM);
    softmax_mask_kernel<<<T_DIM, 256, 0, stream>>>(Sf, cmask, ids, Pb, scale);
    gemm_wmma<1><<<gdim(T_DIM, D_DIM), 256, 0, stream>>>(Pb, Vt, nullptr, AVb,
                                                         T_DIM, D_DIM, S_DIM);
    convT(Wo + (size_t)l * DD, D_DIM, D_DIM);
    gemm_wmma<3><<<gdim(T_DIM, D_DIM), 256, 0, stream>>>(AVb, Wt, x, x,
                                                         T_DIM, D_DIM, D_DIM);

    // ---- cross attention ----
    ln_bf16_kernel<<<T_DIM, 256, 0, stream>>>(x, hbf);
    convT(Cq + (size_t)l * DD, D_DIM, D_DIM);
    gemm_wmma<1><<<gdim(T_DIM, D_DIM), 256, 0, stream>>>(hbf, Wt, nullptr, Qb,
                                                         T_DIM, D_DIM, D_DIM);
    convT(Ck + (size_t)l * DD, D_DIM, D_DIM);
    gemm_wmma<1><<<gdim(S_DIM, D_DIM), 256, 0, stream>>>(encb, Wt, nullptr, Kb,
                                                         S_DIM, D_DIM, D_DIM);
    convT(Cv + (size_t)l * DD, D_DIM, D_DIM);
    gemm_wmma<4><<<gdim(S_DIM, D_DIM), 256, 0, stream>>>(encb, Wt, nullptr, Vt,
                                                         S_DIM, D_DIM, D_DIM);
    gemm_wmma<0><<<gdim(T_DIM, S_DIM), 256, 0, stream>>>(Qb, Kb, nullptr, Sf,
                                                         T_DIM, S_DIM, D_DIM);
    softmax_mask_kernel<<<T_DIM, 256, 0, stream>>>(Sf, cmask, nullptr, Pb, scale);
    gemm_wmma<1><<<gdim(T_DIM, D_DIM), 256, 0, stream>>>(Pb, Vt, nullptr, AVb,
                                                         T_DIM, D_DIM, S_DIM);
    convT(Co + (size_t)l * DD, D_DIM, D_DIM);
    gemm_wmma<3><<<gdim(T_DIM, D_DIM), 256, 0, stream>>>(AVb, Wt, x, x,
                                                         T_DIM, D_DIM, D_DIM);

    // ---- FFN ----
    ln_bf16_kernel<<<T_DIM, 256, 0, stream>>>(x, hbf);
    convT(W1 + (size_t)l * DF, D_DIM, F_DIM);   // -> Wt [F][D]
    gemm_wmma<2><<<gdim(T_DIM, F_DIM), 256, 0, stream>>>(hbf, Wt, nullptr, Hf,
                                                         T_DIM, F_DIM, D_DIM);
    convT(W2 + (size_t)l * DF, F_DIM, D_DIM);   // -> Wt [D][F]
    gemm_wmma<3><<<gdim(T_DIM, D_DIM), 256, 0, stream>>>(Hf, Wt, x, x,
                                                         T_DIM, D_DIM, F_DIM);
  }

  final_kernel<<<T_DIM, 256, 0, stream>>>(x, alpha, gamma, (float*)d_out);
}
